// MultiHeadAttention_23888608101125
// MI455X (gfx1250) — matrix operand order repaired and checked by reading, not compiled, not run on hardware
//
#include <hip/hip_runtime.h>
#include <cmath>

// ---------------------------------------------------------------------------
// MI455X (gfx1250) sparse multi-head attention, bf16 WMMA pipeline.
// B=2, S=2048, E=1024, H=16, D=64, top-k = 256.
// Compute-bound: ~100 GFLOP vs ~50MB HBM traffic -> bf16 WMMA everywhere,
// fp32 scores in LDS for top-k/softmax.  TDM stages GEMM A-tiles into LDS.
// ---------------------------------------------------------------------------

typedef __bf16 bf16_t;
typedef __attribute__((ext_vector_type(16))) __bf16 v16bf;
typedef __attribute__((ext_vector_type(8)))  float  v8f;
typedef unsigned int u32x4 __attribute__((ext_vector_type(4)));
typedef int          i32x8 __attribute__((ext_vector_type(8)));
typedef int          i32x4 __attribute__((ext_vector_type(4)));

#define WMMA_BF16(a, b, c) \
    __builtin_amdgcn_wmma_f32_16x16x32_bf16(false, (a), false, (b), (short)0, (c), false, false)

#if defined(__has_builtin)
#  if __has_builtin(__builtin_amdgcn_tensor_load_to_lds)
#    define HAS_TDM 1
#  endif
#endif

__device__ inline void wait_tensorcnt0() {
#if defined(__has_builtin)
#  if __has_builtin(__builtin_amdgcn_s_wait_tensorcnt)
    __builtin_amdgcn_s_wait_tensorcnt(0);
    return;
#  endif
#endif
    asm volatile("s_wait_tensorcnt 0x0" ::: "memory");
}

static constexpr int Bb = 2;
static constexpr int Ss = 2048;
static constexpr int Ee = 1024;
static constexpr int Hh = 16;
static constexpr int Dd = 64;
static constexpr int NROWS = Bb * Ss;      // 4096
static constexpr int TOPK  = 256;

// ---------------------------------------------------------------------------
// Fragment loaders (wave32; ISA 7.12.2 layouts)
// ---------------------------------------------------------------------------
// 16-bit A matrix 16x32 (MxK): lane m / m+16 holds K = 8h+[0..7], 16+8h+[0..7].
template <typename PT>
__device__ inline v16bf load_a_frag(const PT* base, int ld) {
    int lane = threadIdx.x & 31;
    int m = lane & 15, half = lane >> 4;
    const PT* p = base + m * ld;
    v16bf a;
#pragma unroll
    for (int e = 0; e < 8; ++e) a[e] = (__bf16)p[half * 8 + e];
#pragma unroll
    for (int e = 0; e < 8; ++e) a[8 + e] = (__bf16)p[16 + half * 8 + e];
    return a;
}

// 16-bit B matrix 32x16 (KxN): lane n holds column n, K-contiguous in source.
__device__ inline v16bf load_b_frag(const bf16_t* base, int ld) {
    int lane = threadIdx.x & 31;
    int n = lane & 15, half = lane >> 4;
    const bf16_t* p = base + (size_t)n * ld;
    v16bf b;
#pragma unroll
    for (int e = 0; e < 8; ++e) b[e] = p[half * 8 + e];
#pragma unroll
    for (int e = 0; e < 8; ++e) b[8 + e] = p[16 + half * 8 + e];
    return b;
}

// ---------------------------------------------------------------------------
// Kernel 0: fp32 -> bf16 conversion
// ---------------------------------------------------------------------------
__global__ void cvt_bf16_kernel(const float* __restrict__ in,
                                bf16_t* __restrict__ out, int n) {
    int i = blockIdx.x * blockDim.x + threadIdx.x;
    if (i < n) out[i] = (__bf16)in[i];
}

// ---------------------------------------------------------------------------
// TDM: stage a 32x1024 bf16 tile (64KB) from global into LDS, one descriptor.
// Descriptor per ISA cdna5 §8: group0 = {count, lds_addr, global_addr, type=2},
// group1 = {data_size=2B, dims, tile dims, strides}, groups 2/3 = 0 (2D).
// ---------------------------------------------------------------------------
__device__ inline void tdm_load_tile_32x1024(const bf16_t* gsrc, bf16_t* ldst) {
#if defined(HAS_TDM)
    unsigned lds_off = (unsigned)(size_t)(void*)ldst;      // low 32 = LDS byte addr
    unsigned long long ga = (unsigned long long)(const void*)gsrc;
    u32x4 g0;
    g0[0] = 1u;                                            // count=1, is_restore=0
    g0[1] = lds_off;
    g0[2] = (unsigned)(ga & 0xffffffffu);
    g0[3] = (unsigned)((ga >> 32) & 0x1ffffffu) | (2u << 30);  // type=2 (image)
    i32x8 g1;
    g1[0] = (1 << 16);                       // data_size=1 (2 bytes)
    g1[1] = (int)((1024u & 0xffffu) << 16);  // tensor_dim0[15:0]=1024
    g1[2] = (int)((4096u & 0xffffu) << 16);  // dim0[31:16]=0 | tensor_dim1[15:0]=4096
    g1[3] = (int)(1024u << 16);              // dim1[31:16]=0 | tile_dim0=1024
    g1[4] = 32;                              // tile_dim1=32, tile_dim2=0
    g1[5] = 1024;                            // tensor_dim0_stride[31:0]
    g1[6] = 0;                               // stride0[47:32] | stride1[15:0]
    g1[7] = 0;
    i32x4 gz = {0, 0, 0, 0};
#  if (__clang_major__ <= 22)
    __builtin_amdgcn_tensor_load_to_lds(g0, g1, gz, gz, 0);
#  else
    i32x8 gz8 = {0, 0, 0, 0, 0, 0, 0, 0};
    __builtin_amdgcn_tensor_load_to_lds(g0, g1, gz, gz, gz8, 0);
#  endif
    wait_tensorcnt0();
#else
    // fallback: cooperative 16B copies (whole workgroup, caller barriers)
    const uint4* src = (const uint4*)gsrc;
    uint4* dst = (uint4*)ldst;
    for (int i = threadIdx.x; i < 32 * 1024 * 2 / 16; i += 256) dst[i] = src[i];
#endif
}

// ---------------------------------------------------------------------------
// Kernel 1/3: C = A @ W^T + bias.  A: NROWS x 1024 bf16, W: 1024x1024 bf16.
// 8 waves/WG; wave tile M16 x N64; WG tile M32 x N256.  A tile staged to LDS
// by TDM; W fragments software-pipelined from global (L2-resident).
// layout 0: fp32 row-major   1: bf16 (B,H,S,D)   2: bf16 (B,H,D,S) transposed
// ---------------------------------------------------------------------------
__global__ __launch_bounds__(256) void gemm_bf16_kernel(
    const bf16_t* __restrict__ A, const bf16_t* __restrict__ W,
    const float* __restrict__ bias, bf16_t* __restrict__ Cbf,
    float* __restrict__ Cf, int layout) {
    extern __shared__ char gsm[];
    bf16_t* sA = (bf16_t*)gsm;                  // 32 x 1024 bf16 = 64KB

    int w    = threadIdx.x >> 5;
    int lane = threadIdx.x & 31;
    int wrow = w >> 2;              // 0..1
    int wcol = w & 3;               // 0..3
    int m0 = blockIdx.y * 32 + wrow * 16;
    int n0 = blockIdx.x * 256 + wcol * 64;

#if defined(HAS_TDM)
    if (w == 0) tdm_load_tile_32x1024(A + (size_t)(blockIdx.y * 32) * Ee, sA);
#else
    tdm_load_tile_32x1024(A + (size_t)(blockIdx.y * 32) * Ee, sA);
#endif
    __syncthreads();

    const bf16_t* Arow = sA + wrow * 16 * Ee;   // LDS, row stride 1024

    v8f acc[4] = {};
    // software pipeline: preload k=0 fragments
    v16bf a_cur = load_a_frag(Arow, Ee);
    v16bf b0 = load_b_frag(W + (size_t)(n0 +  0) * Ee, Ee);
    v16bf b1 = load_b_frag(W + (size_t)(n0 + 16) * Ee, Ee);
    v16bf b2 = load_b_frag(W + (size_t)(n0 + 32) * Ee, Ee);
    v16bf b3 = load_b_frag(W + (size_t)(n0 + 48) * Ee, Ee);
    for (int kk = 0; kk < Ee; kk += 32) {
        int kn = (kk + 32 < Ee) ? kk + 32 : kk;
        __builtin_prefetch(W + (size_t)(n0 + (lane & 15)) * Ee + kk + 128, 0, 1);
        v16bf a_n = load_a_frag(Arow + kn, Ee);
        v16bf c0 = load_b_frag(W + (size_t)(n0 +  0) * Ee + kn, Ee);
        v16bf c1 = load_b_frag(W + (size_t)(n0 + 16) * Ee + kn, Ee);
        v16bf c2 = load_b_frag(W + (size_t)(n0 + 32) * Ee + kn, Ee);
        v16bf c3 = load_b_frag(W + (size_t)(n0 + 48) * Ee + kn, Ee);
        acc[0] = WMMA_BF16(a_cur, b0, acc[0]);
        acc[1] = WMMA_BF16(a_cur, b1, acc[1]);
        acc[2] = WMMA_BF16(a_cur, b2, acc[2]);
        acc[3] = WMMA_BF16(a_cur, b3, acc[3]);
        a_cur = a_n; b0 = c0; b1 = c1; b2 = c2; b3 = c3;
    }

    int nl = lane & 15, half = lane >> 4;
#pragma unroll
    for (int t = 0; t < 4; ++t) {
#pragma unroll
        for (int r = 0; r < 8; ++r) {
            int m = r + 8 * half;
            int i = m0 + m;
            int j = n0 + t * 16 + nl;
            float v = acc[t][r] + bias[j];
            if (layout == 1) {            // (B,H,S,D)
                int b = i >> 11, s = i & (Ss - 1);
                int h = j >> 6,  d = j & (Dd - 1);
                Cbf[((size_t)(b * Hh + h) * Ss + s) * Dd + d] = (__bf16)v;
            } else if (layout == 2) {     // (B,H,D,S)  (V transposed)
                int b = i >> 11, s = i & (Ss - 1);
                int h = j >> 6,  d = j & (Dd - 1);
                Cbf[((size_t)(b * Hh + h) * Dd + d) * Ss + s] = (__bf16)v;
            } else {
                Cf[(size_t)i * Ee + j] = v;
            }
        }
    }
}

// ---------------------------------------------------------------------------
// Kernel 2: sparse attention.  One WG per (b, h, 16-query tile).
// 8 waves; wave w owns T slice [w*256, w*256+256).
// ---------------------------------------------------------------------------
static constexpr size_t ATTN_LDS =
    (size_t)16 * Ss * 4 + (size_t)16 * Ss * 2 + (size_t)8 * 16 * Dd * 4
    + 3 * 16 * 4 + (size_t)16 * Dd * 2;

__global__ __launch_bounds__(256) void attn_topk_kernel(
    const bf16_t* __restrict__ Q, const bf16_t* __restrict__ K,
    const bf16_t* __restrict__ VT, bf16_t* __restrict__ O) {
    extern __shared__ char smem[];
    float*  s_scores = (float*)smem;                                   // 16*2048 f32
    bf16_t* s_attn   = (bf16_t*)(smem + (size_t)16 * Ss * 4);          // 16*2048 bf16
    float*  s_red    = (float*)(smem + (size_t)16 * Ss * 4 + (size_t)16 * Ss * 2);
    float*  s_thr    = s_red + 8 * 16 * Dd;
    float*  s_max    = s_thr + 16;
    float*  s_inv    = s_max + 16;
    bf16_t* s_q      = (bf16_t*)(s_inv + 16);                          // 16*64 bf16

    int w    = threadIdx.x >> 5;
    int lane = threadIdx.x & 31;
    int q0   = blockIdx.x * 16;
    int bh   = blockIdx.z * Hh + blockIdx.y;

    const bf16_t* Qb  = Q  + ((size_t)bh * Ss + q0) * Dd;
    const bf16_t* Kb  = K  + (size_t)bh * Ss * Dd;
    const bf16_t* VTb = VT + (size_t)bh * Dd * Ss;
    int tbase = w * 256;

    // ---- stage Q tile (16x64 bf16 = 2KB) into LDS ------------------------
    {
        const uint4* src = (const uint4*)Qb;
        uint4* dst = (uint4*)s_q;
        for (int i = threadIdx.x; i < 16 * Dd * 2 / 16; i += 256) dst[i] = src[i];
    }
    __syncthreads();

    // ---- Phase 1: scores = (Q K^T) / sqrt(D), pipelined ------------------
    v16bf aq0 = load_a_frag(s_q + 0,  Dd);
    v16bf aq1 = load_a_frag(s_q + 32, Dd);
    int nl = lane & 15, half = lane >> 4;
    v16bf kb0 = load_b_frag(Kb + (size_t)tbase * Dd + 0,  Dd);
    v16bf kb1 = load_b_frag(Kb + (size_t)tbase * Dd + 32, Dd);
    for (int nt = 0; nt < 16; ++nt) {
        int t0 = tbase + nt * 16;
        int tn = (nt < 15) ? t0 + 16 : t0;
        v16bf nb0 = load_b_frag(Kb + (size_t)tn * Dd + 0,  Dd);
        v16bf nb1 = load_b_frag(Kb + (size_t)tn * Dd + 32, Dd);
        v8f acc = {};
        acc = WMMA_BF16(aq0, kb0, acc);
        acc = WMMA_BF16(aq1, kb1, acc);
#pragma unroll
        for (int r = 0; r < 8; ++r) {
            int m = r + 8 * half;
            s_scores[m * Ss + t0 + nl] = acc[r] * 0.125f;  // 1/sqrt(64)
        }
        kb0 = nb0; kb1 = nb1;
    }
    __syncthreads();

    // ---- Phase 2: per-row top-256 threshold (bisection) + softmax stats --
    for (int rr = 0; rr < 2; ++rr) {
        int row = w * 2 + rr;
        const float* sr = s_scores + row * Ss;
        float mx = -3.4e38f, mn = 3.4e38f;
        for (int i = lane; i < Ss; i += 32) {
            float v = sr[i];
            mx = fmaxf(mx, v);
            mn = fminf(mn, v);
        }
#pragma unroll
        for (int off = 16; off > 0; off >>= 1) {
            mx = fmaxf(mx, __shfl_xor(mx, off, 32));
            mn = fminf(mn, __shfl_xor(mn, off, 32));
        }
        float lo = mn, hi = mx;
        for (int it = 0; it < 24; ++it) {
            float mid = 0.5f * (lo + hi);
            int cnt = 0;
            for (int i = lane; i < Ss; i += 32) cnt += (sr[i] >= mid) ? 1 : 0;
#pragma unroll
            for (int off = 16; off > 0; off >>= 1) cnt += __shfl_xor(cnt, off, 32);
            if (cnt >= TOPK) lo = mid; else hi = mid;
        }
        float thr = lo;
        float sum = 0.f;
        for (int i = lane; i < Ss; i += 32) {
            float v = sr[i];
            if (v >= thr) sum += __expf(v - mx);
        }
#pragma unroll
        for (int off = 16; off > 0; off >>= 1) sum += __shfl_xor(sum, off, 32);
        if (lane == 0) {
            s_thr[row] = thr;
            s_max[row] = mx;
            s_inv[row] = 1.f / sum;
        }
    }
    __syncthreads();

    // ---- Phase 3: masked softmax -> bf16 attention weights ---------------
    for (int row = 0; row < 16; ++row) {
        float thr = s_thr[row], mx = s_max[row], inv = s_inv[row];
        for (int i = tbase + lane; i < tbase + 256; i += 32) {
            float v = s_scores[row * Ss + i];
            float a = (v >= thr) ? __expf(v - mx) * inv : 0.f;
            s_attn[row * Ss + i] = (__bf16)a;
        }
    }
    __syncthreads();

    // ---- Phase 4: out = attn @ V  (V stored transposed: contiguous B) ----
    v8f acc[4] = {};
    v16bf vb0 = load_b_frag(VTb + (size_t) 0 * Ss + tbase, Ss);
    v16bf vb1 = load_b_frag(VTb + (size_t)16 * Ss + tbase, Ss);
    v16bf vb2 = load_b_frag(VTb + (size_t)32 * Ss + tbase, Ss);
    v16bf vb3 = load_b_frag(VTb + (size_t)48 * Ss + tbase, Ss);
    for (int t0 = tbase; t0 < tbase + 256; t0 += 32) {
        int tn = (t0 + 32 < tbase + 256) ? t0 + 32 : t0;
        v16bf n0 = load_b_frag(VTb + (size_t) 0 * Ss + tn, Ss);
        v16bf n1 = load_b_frag(VTb + (size_t)16 * Ss + tn, Ss);
        v16bf n2 = load_b_frag(VTb + (size_t)32 * Ss + tn, Ss);
        v16bf n3 = load_b_frag(VTb + (size_t)48 * Ss + tn, Ss);
        v16bf a = load_a_frag(s_attn + t0, Ss);
        acc[0] = WMMA_BF16(a, vb0, acc[0]);
        acc[1] = WMMA_BF16(a, vb1, acc[1]);
        acc[2] = WMMA_BF16(a, vb2, acc[2]);
        acc[3] = WMMA_BF16(a, vb3, acc[3]);
        vb0 = n0; vb1 = n1; vb2 = n2; vb3 = n3;
    }
#pragma unroll
    for (int dt = 0; dt < 4; ++dt)
#pragma unroll
        for (int r = 0; r < 8; ++r) {
            int m = r + 8 * half;
            s_red[(w * 16 + m) * Dd + dt * 16 + nl] = acc[dt][r];
        }
    __syncthreads();

    // cross-wave reduction and store to (B,S,E) with head offset
    for (int idx = threadIdx.x; idx < 16 * Dd; idx += 256) {
        int m = idx >> 6, d = idx & (Dd - 1);
        float s = 0.f;
#pragma unroll
        for (int ww = 0; ww < 8; ++ww) s += s_red[(ww * 16 + m) * Dd + d];
        int b = blockIdx.z, h = blockIdx.y;
        O[((size_t)(b * Ss + q0 + m) * Ee) + h * Dd + d] = (__bf16)s;
    }
}

// ---------------------------------------------------------------------------
// Host launcher
// ---------------------------------------------------------------------------
extern "C" void kernel_launch(void* const* d_in, const int* in_sizes, int n_in,
                              void* d_out, int out_size, void* d_ws, size_t ws_size,
                              hipStream_t stream) {
    (void)in_sizes; (void)n_in; (void)out_size; (void)ws_size;

    const float* x  = (const float*)d_in[0];
    const float* Wq = (const float*)d_in[1];
    const float* bq = (const float*)d_in[2];
    const float* Wk = (const float*)d_in[3];
    const float* bk = (const float*)d_in[4];
    const float* Wv = (const float*)d_in[5];
    const float* bv = (const float*)d_in[6];
    const float* Wo = (const float*)d_in[7];
    const float* bo = (const float*)d_in[8];
    float* out = (float*)d_out;

    char* ws = (char*)d_ws;
    bf16_t* xbf  = (bf16_t*)(ws + 0);                         // 4096x1024 = 8MB
    bf16_t* Wqbf = (bf16_t*)(ws + (8u << 20));
    bf16_t* Wkbf = (bf16_t*)(ws + (10u << 20));
    bf16_t* Wvbf = (bf16_t*)(ws + (12u << 20));
    bf16_t* Wobf = (bf16_t*)(ws + (14u << 20));
    bf16_t* Qb   = (bf16_t*)(ws + (16u << 20));               // (B,H,S,D) 8MB
    bf16_t* Kb   = (bf16_t*)(ws + (24u << 20));
    bf16_t* VTb  = (bf16_t*)(ws + (32u << 20));               // (B,H,D,S) 8MB
    bf16_t* AO   = (bf16_t*)(ws + (40u << 20));               // (B,S,E) 8MB

    const int nx = NROWS * Ee;
    const int nw = Ee * Ee;
    cvt_bf16_kernel<<<(nx + 255) / 256, 256, 0, stream>>>(x,  xbf,  nx);
    cvt_bf16_kernel<<<(nw + 255) / 256, 256, 0, stream>>>(Wq, Wqbf, nw);
    cvt_bf16_kernel<<<(nw + 255) / 256, 256, 0, stream>>>(Wk, Wkbf, nw);
    cvt_bf16_kernel<<<(nw + 255) / 256, 256, 0, stream>>>(Wv, Wvbf, nw);
    cvt_bf16_kernel<<<(nw + 255) / 256, 256, 0, stream>>>(Wo, Wobf, nw);

    const size_t GEMM_LDS = (size_t)32 * Ee * 2;   // 64KB A-tile
    (void)hipFuncSetAttribute((const void*)gemm_bf16_kernel,
                              hipFuncAttributeMaxDynamicSharedMemorySize, (int)GEMM_LDS);
    dim3 ggrid(Ee / 256, NROWS / 32);   // (4, 128)
    gemm_bf16_kernel<<<ggrid, 256, GEMM_LDS, stream>>>(xbf, Wqbf, bq, Qb,  nullptr, 1);
    gemm_bf16_kernel<<<ggrid, 256, GEMM_LDS, stream>>>(xbf, Wkbf, bk, Kb,  nullptr, 1);
    gemm_bf16_kernel<<<ggrid, 256, GEMM_LDS, stream>>>(xbf, Wvbf, bv, VTb, nullptr, 2);

    (void)hipFuncSetAttribute((const void*)attn_topk_kernel,
                              hipFuncAttributeMaxDynamicSharedMemorySize, (int)ATTN_LDS);
    dim3 agrid(Ss / 16, Hh, Bb);        // (128, 16, 2)
    attn_topk_kernel<<<agrid, 256, ATTN_LDS, stream>>>(Qb, Kb, VTb, AO);

    gemm_bf16_kernel<<<ggrid, 256, GEMM_LDS, stream>>>(AO, Wobf, bo, nullptr, out, 0);
}
